// GCN_40020505264508
// MI455X (gfx1250) — matrix operand order repaired and checked by reading, not compiled, not run on hardware
//
#include <hip/hip_runtime.h>
#include <stdint.h>

#define N_NODES 100000
#define DIN 128

typedef __attribute__((ext_vector_type(16))) __bf16 v16bf;
typedef __attribute__((ext_vector_type(8)))  float  v8f;

__device__ __forceinline__ unsigned short f32_to_bf16_bits(float f) {
  union { float f; unsigned int u; } c; c.f = f;
  unsigned int u = c.u;
  unsigned int r = u + 0x7FFFu + ((u >> 16) & 1u);  // round-to-nearest-even
  return (unsigned short)(r >> 16);
}

__global__ void zero_f32(float* __restrict__ p, long long n) {
  long long i = (long long)blockIdx.x * blockDim.x + threadIdx.x;
  long long stride = (long long)gridDim.x * blockDim.x;
  for (; i < n; i += stride) p[i] = 0.0f;
}

// Convert + swizzle a weight matrix (DIN x dout, f32 row-major) into WMMA B-operand
// tile order: outB[((kt*tilesN + nt)*32 + lane)*16 + e]  with
//   col = nt*16 + (lane&15),  k = kt*32 + (e<8 ? e : e+8) + 8*(lane>>4)
__global__ void pack_b(const float* __restrict__ W, unsigned short* __restrict__ outB, int dout) {
  int i = blockIdx.x * blockDim.x + threadIdx.x;
  if (i >= DIN * dout) return;
  int tilesN = dout >> 4;
  int e = i & 15;
  int lane = (i >> 4) & 31;
  int tileIdx = i >> 9;
  int nt = tileIdx % tilesN;
  int kt = tileIdx / tilesN;
  int hi = lane >> 4;
  int k = kt * 32 + (e < 8 ? e : e + 8) + 8 * hi;
  int col = nt * 16 + (lane & 15);
  outB[i] = f32_to_bf16_bits(W[(size_t)k * dout + col]);
}

// Convert activations (f32 row-major [N][128], optional per-row 1/deg scaling) into
// WMMA A-operand tile order: outA[((mtile*4 + kt)*32 + lane)*16 + e] with
//   node = mtile*16 + (lane&15),  k = kt*32 + (e<8 ? e : e+8) + 8*(lane>>4)
__global__ void pack_a(const float* __restrict__ in, const float* __restrict__ deg,
                       unsigned short* __restrict__ outA) {
  int i = blockIdx.x * blockDim.x + threadIdx.x;
  if (i >= N_NODES * DIN) return;
  int e = i & 15;
  int lane = (i >> 4) & 31;
  int kt = (i >> 9) & 3;
  int mtile = i >> 11;
  int hi = lane >> 4;
  int node = mtile * 16 + (lane & 15);
  int k = kt * 32 + (e < 8 ? e : e + 8) + 8 * hi;
  float v = in[(size_t)node * DIN + k];
  if (deg) v *= 1.0f / fmaxf(deg[node], 1.0f);
  outA[i] = f32_to_bf16_bits(v);
}

// One wave per edge: 32 lanes x float4 = 128 features, coalesced 512B row read,
// scatter-add via global_atomic_add_f32 (both tables L2-resident: 51MB < 192MB L2).
__global__ void scatter_edges(const float* __restrict__ x,
                              const long long* __restrict__ src,
                              const long long* __restrict__ dst,
                              float* __restrict__ agg, float* __restrict__ deg,
                              int n_edges) {
  int gtid = blockIdx.x * blockDim.x + threadIdx.x;
  int wave = gtid >> 5;
  int lane = threadIdx.x & 31;
  int nwaves = (gridDim.x * blockDim.x) >> 5;
  for (int e = wave; e < n_edges; e += nwaves) {
    int s = (int)src[e];
    int d = (int)dst[e];
    const float4 v = *(const float4*)(x + (size_t)s * DIN + lane * 4);
    float* a = agg + (size_t)d * DIN + lane * 4;
    atomicAdd(a + 0, v.x);
    atomicAdd(a + 1, v.y);
    atomicAdd(a + 2, v.z);
    atomicAdd(a + 3, v.w);
    if (lane == 0) atomicAdd(deg + d, 1.0f);
  }
}

// Fused dense layer on pre-packed bf16 operands:
//   out = mean @ Wl + bl + x @ Wr, one wave -> one 16x16 f32 tile.
// Inner loop: 8 coalesced global_load_b128 + 2 v_wmma per 32-wide K step.
__global__ void sage_gemm(const unsigned short* __restrict__ meanA,
                          const unsigned short* __restrict__ xA,
                          const unsigned short* __restrict__ wl,
                          const unsigned short* __restrict__ wr,
                          const float* __restrict__ bias,
                          float* __restrict__ out, int dout) {
  const int lane = threadIdx.x & 31;
  const int tilesN = dout >> 4;
  int tile = blockIdx.x * (blockDim.x >> 5) + (threadIdx.x >> 5);
  const int mtile = tile / tilesN;
  const int ntile = tile % tilesN;
  if (mtile >= N_NODES / 16) return;  // wave-uniform; grid is exact so never taken

  union U { v16bf v; int4 q[2]; };
  v8f c = {};
#pragma unroll
  for (int kt = 0; kt < 4; ++kt) {
    const size_t aoff = ((size_t)(mtile * 4 + kt) * 32 + lane) * 16;
    const size_t boff = ((size_t)(kt * tilesN + ntile) * 32 + lane) * 16;
    const int4* pa = (const int4*)(meanA + aoff);
    const int4* px = (const int4*)(xA + aoff);
    const int4* pl = (const int4*)(wl + boff);
    const int4* pr = (const int4*)(wr + boff);
    U Am, Ax, Bl, Br;
    Am.q[0] = pa[0]; Am.q[1] = pa[1];
    Ax.q[0] = px[0]; Ax.q[1] = px[1];
    Bl.q[0] = pl[0]; Bl.q[1] = pl[1];
    Br.q[0] = pr[0]; Br.q[1] = pr[1];
    c = __builtin_amdgcn_wmma_f32_16x16x32_bf16(false, Am.v, false, Bl.v, (short)0, c, false, false);
    c = __builtin_amdgcn_wmma_f32_16x16x32_bf16(false, Ax.v, false, Br.v, (short)0, c, false, false);
  }

  // C layout: VGPR i holds (M = i + 8*(lane>>4), N = lane&15)
  const int hi = lane >> 4;
  const int col = ntile * 16 + (lane & 15);
  const float b = bias[col];
#pragma unroll
  for (int i = 0; i < 8; ++i)
    out[(size_t)(mtile * 16 + i + 8 * hi) * dout + col] = c[i] + b;
}

// L2-normalize each node row (+ optional ReLU). One wave per node, shfl_xor reduce (wave32).
__global__ void l2norm_relu(const float* __restrict__ in, float* __restrict__ out,
                            int d, int do_relu) {
  int wave = (blockIdx.x * blockDim.x + threadIdx.x) >> 5;
  int lane = threadIdx.x & 31;
  int nwaves = (gridDim.x * blockDim.x) >> 5;
  int per = d >> 5;  // 4 (d=128) or 2 (d=64)
  for (int n = wave; n < N_NODES; n += nwaves) {
    const float* r = in + (size_t)n * d;
    float v[4];
    float ss = 0.0f;
    for (int j = 0; j < per; ++j) { v[j] = r[lane + j * 32]; ss += v[j] * v[j]; }
    for (int off = 16; off > 0; off >>= 1) ss += __shfl_xor(ss, off, 32);
    float sc = 1.0f / fmaxf(sqrtf(ss), 1e-12f);
    float* o = out + (size_t)n * d;
    for (int j = 0; j < per; ++j) {
      float y = v[j] * sc;
      if (do_relu) y = fmaxf(y, 0.0f);
      o[lane + j * 32] = y;
    }
  }
}

extern "C" void kernel_launch(void* const* d_in, const int* in_sizes, int n_in,
                              void* d_out, int out_size, void* d_ws, size_t ws_size,
                              hipStream_t stream) {
  const float*     x  = (const float*)d_in[0];
  const long long* ei = (const long long*)d_in[1];  // jnp.int64, shape (2, E) row-major
  const int n_edges = in_sizes[1] / 2;
  const long long* srcp = ei;
  const long long* dstp = ei + n_edges;

  const float* Wl[3] = {(const float*)d_in[2], (const float*)d_in[5], (const float*)d_in[8]};
  const float* bl[3] = {(const float*)d_in[3], (const float*)d_in[6], (const float*)d_in[9]};
  const float* Wr[3] = {(const float*)d_in[4], (const float*)d_in[7], (const float*)d_in[10]};
  const int wsz[3] = {128 * 128, 128 * 128, 128 * 64};
  const int douts[3] = {128, 128, 64};

  // ---- workspace layout (all 16B-aligned by construction)
  float* agg = (float*)d_ws;                               // N*128 f32
  float* deg = agg + (size_t)N_NODES * DIN;                // N f32
  float* act = deg + N_NODES;                              // N*128 f32 (in-place activations)
  unsigned short* meanA = (unsigned short*)(act + (size_t)N_NODES * DIN);  // N*128 bf16
  unsigned short* xA    = meanA + (size_t)N_NODES * DIN;                   // N*128 bf16
  unsigned short* wbuf  = xA + (size_t)N_NODES * DIN;
  unsigned short* wlb[3], *wrb[3];
  {
    unsigned short* p = wbuf;
    for (int i = 0; i < 3; ++i) { wlb[i] = p; p += wsz[i]; wrb[i] = p; p += wsz[i]; }
  }

  // ---- weights: convert + swizzle into WMMA B layout (tiny, L2/L0-resident afterwards)
  for (int i = 0; i < 3; ++i) {
    pack_b<<<(wsz[i] + 255) / 256, 256, 0, stream>>>(Wl[i], wlb[i], douts[i]);
    pack_b<<<(wsz[i] + 255) / 256, 256, 0, stream>>>(Wr[i], wrb[i], douts[i]);
  }

  const int packGrid = (N_NODES * DIN + 255) / 256;

  auto run_layer = [&](const float* xin, float* norm_out, int layer, int dout, bool relu) {
    zero_f32<<<2048, 256, 0, stream>>>(agg, (long long)N_NODES * DIN);
    zero_f32<<<256, 256, 0, stream>>>(deg, (long long)N_NODES);
    scatter_edges<<<8192, 256, 0, stream>>>(xin, srcp, dstp, agg, deg, n_edges);
    pack_a<<<packGrid, 256, 0, stream>>>(xin, nullptr, xA);     // root path operand
    pack_a<<<packGrid, 256, 0, stream>>>(agg, deg, meanA);      // mean-aggregated operand
    int tiles = (N_NODES / 16) * (dout / 16);                   // divisible by 8 waves/block
    sage_gemm<<<tiles / 8, 256, 0, stream>>>(meanA, xA, wlb[layer], wrb[layer],
                                             bl[layer], act, dout);
    l2norm_relu<<<(N_NODES * 32 + 255) / 256, 256, 0, stream>>>(act, norm_out, dout,
                                                                relu ? 1 : 0);
  };

  run_layer(x,   act,           0, 128, true);   // gemm overwrites act; norm in-place
  run_layer(act, act,           1, 128, true);
  run_layer(act, (float*)d_out, 2, 64,  false);
}
